// MultiHeadAttentionLayer_10849087390183
// MI455X (gfx1250) — matrix operand order repaired and checked by reading, not compiled, not run on hardware
//
#include <hip/hip_runtime.h>

#define DEV __device__ __forceinline__

typedef __bf16 bf16;
typedef __attribute__((ext_vector_type(16))) __bf16 v16bf;
typedef __attribute__((ext_vector_type(8)))  float  v8f;
typedef __attribute__((ext_vector_type(4)))  int    v4i;

constexpr int Bb = 4, Ss = 2048, Dd = 1024, Hh = 16, DHh = 64;
constexpr int MROWS = Bb * Ss;  // 8192

// ---------------------------------------------------------------------------
// Async global->LDS (CDNA5: GLOBAL_LOAD_ASYNC_TO_LDS_B128, ASYNCcnt).
// Builtin signature (from clang diagnostic): param0 = int4 AS(1)*, then
// LDS dst, imm offset, imm cpol.
// ---------------------------------------------------------------------------
#if defined(__has_builtin)
#if __has_builtin(__builtin_amdgcn_global_load_async_to_lds_b128) && \
    __has_builtin(__builtin_amdgcn_s_wait_asynccnt)
#define ASYNC_LDS 1
#endif
#endif
#ifndef ASYNC_LDS
#define ASYNC_LDS 0
#endif

DEV void cp16(bf16* lds, const bf16* g) {
#if ASYNC_LDS
  __builtin_amdgcn_global_load_async_to_lds_b128(
      (v4i __attribute__((address_space(1)))*)g,
      (v4i __attribute__((address_space(3)))*)lds, 0, 0);
#else
  *(uint4*)lds = *(const uint4*)g;
#endif
}

#if ASYNC_LDS
#define ASYNC_WAIT(n) __builtin_amdgcn_s_wait_asynccnt(n)
#else
#define ASYNC_WAIT(n)
#endif

// ---------------------------------------------------------------------------
// WMMA helpers (bf16 in, f32 accumulate). CDNA5 16x16x32 layouts:
//  A (16x32): lane l holds row M=l&15; hi=l>>4; K chunks [hi*8,hi*8+8) and
//             [16+hi*8, ...) -> two 16B LDS reads from an [m][k] tile.
//  B (32x16): lane l holds col N=l&15; same K chunks as A -> two 16B
//             reads from an [n][k] tile.
//  C/D (16x16 f32): VGPR r holds row M=(l>>4)*8+r, col N=l&15.
// ---------------------------------------------------------------------------
DEV v8f wmma_bf16(v16bf a, v16bf b, v8f c) {
  return __builtin_amdgcn_wmma_f32_16x16x32_bf16(false, a, false, b,
                                                 (short)0, c, false, false);
}

DEV v16bf load_a_frag(const bf16* base, int ldk, int lane) {
  int l = lane & 15, hi = lane >> 4;
  const bf16* row = base + l * ldk;
  v16bf r;
  ((uint4*)&r)[0] = *(const uint4*)(row + hi * 8);
  ((uint4*)&r)[1] = *(const uint4*)(row + 16 + hi * 8);
  return r;
}

DEV v16bf load_b_frag(const bf16* base, int ldk, int kbase, int lane) {
  int n = lane & 15, hi = lane >> 4;
  const bf16* p = base + n * ldk + kbase;
  v16bf r;
  ((uint4*)&r)[0] = *(const uint4*)(p + hi * 8);
  ((uint4*)&r)[1] = *(const uint4*)(p + 16 + hi * 8);
  return r;
}

// ---------------------------------------------------------------------------
// f32 -> bf16 elementwise convert (n divisible by 4)
// ---------------------------------------------------------------------------
__global__ __launch_bounds__(256)
void cvt_bf16_kernel(const float* __restrict__ in, bf16* __restrict__ out, long n) {
  long i = ((long)blockIdx.x * blockDim.x + threadIdx.x) * 4;
  if (i >= n) return;
  float4 v = *(const float4*)(in + i);
  bf16 r[4] = {(bf16)v.x, (bf16)v.y, (bf16)v.z, (bf16)v.w};
  *(uint2*)(out + i) = *(uint2*)r;
}

// ---------------------------------------------------------------------------
// f32 [K,N] -> bf16 [N,K] transposed convert (weights).
// ---------------------------------------------------------------------------
__global__ __launch_bounds__(256)
void transpose_bf16_kernel(const float* __restrict__ in, bf16* __restrict__ out,
                           int K, int N) {
  __shared__ bf16 tile[64][68];
  int bx = blockIdx.x * 64, by = blockIdx.y * 64;
  int tx = threadIdx.x, ty = threadIdx.y;
#pragma unroll
  for (int i = 0; i < 16; ++i)
    tile[tx][ty * 16 + i] = (bf16)in[(size_t)(by + ty * 16 + i) * N + bx + tx];
  __syncthreads();
#pragma unroll
  for (int i = 0; i < 16; ++i)
    out[(size_t)(bx + ty * 16 + i) * K + by + tx] = tile[ty * 16 + i][tx];
}

// ---------------------------------------------------------------------------
// GEMM: C[M,N] = A[M,K] @ W[K,N] + bias, W supplied transposed Bt=[N,K].
// Block 128 threads = 4 waves, tile 128x128, wave tile 64x64, K-step 32.
// Double-buffered LDS with async global->LDS prefetch of tile j+1 under the
// WMMAs of tile j.
// ---------------------------------------------------------------------------
__global__ __launch_bounds__(128)
void gemm_bf16_kernel(const bf16* __restrict__ A, const bf16* __restrict__ Bt,
                      const float* __restrict__ bias,
                      bf16* __restrict__ outb, float* __restrict__ outf,
                      int Kdim, int Ndim) {
  __shared__ bf16 sA[2][128 * 32];
  __shared__ bf16 sB[2][128 * 32];
  int tid = threadIdx.x, lane = tid & 31, wave = tid >> 5;
  int m0 = blockIdx.y * 128, n0 = blockIdx.x * 128;
  int wm = (wave >> 1) * 64, wn = (wave & 1) * 64;

  auto loadTiles = [&](int k0, int buf) {
#pragma unroll
    for (int u = tid; u < 512; u += 128) {         // 512 uint4 per 8KB tile
      int row = u >> 2, c = (u & 3) * 8;           // 4 uint4 per 32-elem row
      cp16(&sA[buf][row * 32 + c], A + (size_t)(m0 + row) * Kdim + k0 + c);
      cp16(&sB[buf][row * 32 + c], Bt + (size_t)(n0 + row) * Kdim + k0 + c);
    }
  };

  v8f acc[4][4];
#pragma unroll
  for (int i = 0; i < 4; ++i)
#pragma unroll
    for (int j = 0; j < 4; ++j)
#pragma unroll
      for (int r = 0; r < 8; ++r) acc[i][j][r] = 0.0f;

  loadTiles(0, 0);
  int nk = Kdim / 32;
  for (int j = 0; j < nk; ++j) {
    int cur = j & 1;
    __syncthreads();                    // buf[cur^1] readers (iter j-1) done
    if (j + 1 < nk) {
      loadTiles((j + 1) * 32, cur ^ 1); // prefetch under compute
      ASYNC_WAIT(8);                    // tile j landed; tile j+1 in flight
    } else {
      ASYNC_WAIT(0);
    }
    __syncthreads();                    // tile j visible to all waves

    v16bf bfr[4];
#pragma unroll
    for (int jj = 0; jj < 4; ++jj)
      bfr[jj] = load_b_frag(sB[cur] + (wn + jj * 16) * 32, 32, 0, lane);
#pragma unroll
    for (int i = 0; i < 4; ++i) {
      v16bf af = load_a_frag(sA[cur] + (wm + i * 16) * 32, 32, lane);
#pragma unroll
      for (int jj = 0; jj < 4; ++jj)
        acc[i][jj] = wmma_bf16(af, bfr[jj], acc[i][jj]);
    }
  }

  int ln = lane & 15, lh = lane >> 4;
#pragma unroll
  for (int i = 0; i < 4; ++i)
#pragma unroll
    for (int jj = 0; jj < 4; ++jj)
#pragma unroll
      for (int r = 0; r < 8; ++r) {
        int m = m0 + wm + i * 16 + lh * 8 + r;
        int n = n0 + wn + jj * 16 + ln;
        float v = acc[i][jj][r] + bias[n];
        if (outb) outb[(size_t)m * Ndim + n] = (bf16)v;
        else      outf[(size_t)m * Ndim + n] = v;
      }
}

// ---------------------------------------------------------------------------
// Flash attention: grid (S/64, H, B), block 128 (4 waves, 16 q-rows/wave).
// K-tile double-buffered with async prefetch; V transposed into LDS via VGPRs.
// ---------------------------------------------------------------------------
__global__ __launch_bounds__(128)
void flash_attn_kernel(const bf16* __restrict__ Q, const bf16* __restrict__ K,
                       const bf16* __restrict__ V, const float* __restrict__ mask,
                       bf16* __restrict__ ctx) {
  __shared__ bf16 sQ[64 * 64];
  __shared__ bf16 sK[2][64 * 64];     // [kv][dh]  (B layout for Q@K^T)
  __shared__ bf16 sVt[64 * 64];       // [dh][kv]  (B layout for P@V)
  __shared__ bf16 sP[4][16 * 64];     // per-wave P strip, [q][kv]
  __shared__ float sMask[Ss];

  int tid = threadIdx.x, lane = tid & 31, wave = tid >> 5;
  int ln = lane & 15, lh = lane >> 4;
  int qt = blockIdx.x, h = blockIdx.y, b = blockIdx.z;
  size_t rowbase = (size_t)b * Ss;

  const bf16* Qg = Q + (rowbase + qt * 64) * Dd + h * DHh;
  const bf16* Kg = K + rowbase * Dd + h * DHh;
  const bf16* Vg = V + rowbase * Dd + h * DHh;

  auto kload = [&](int kv0, int buf) {
#pragma unroll
    for (int u = tid; u < 512; u += 128) {
      int r = u >> 3, c = (u & 7) * 8;
      cp16(&sK[buf][r * 64 + c], Kg + (size_t)(kv0 + r) * Dd + c);
    }
  };

  // prologue: Q tile + K(0) async; mask bias
#pragma unroll
  for (int u = tid; u < 512; u += 128) {
    int r = u >> 3, c = (u & 7) * 8;
    cp16(&sQ[r * 64 + c], Qg + (size_t)r * Dd + c);
  }
  kload(0, 0);
  for (int i = tid; i < Ss; i += 128) sMask[i] = -1e9f * mask[b * Ss + i];
  ASYNC_WAIT(0);
  __syncthreads();

  v16bf qa0 = load_a_frag(sQ + wave * 16 * 64, 64, lane);       // dh 0..31
  v16bf qa1 = load_a_frag(sQ + wave * 16 * 64 + 32, 64, lane);  // dh 32..63

  v8f o[4];
  float mrow[8], lrow[8];
#pragma unroll
  for (int j = 0; j < 4; ++j)
#pragma unroll
    for (int r = 0; r < 8; ++r) o[j][r] = 0.0f;
#pragma unroll
  for (int r = 0; r < 8; ++r) { mrow[r] = -3.0e38f; lrow[r] = 0.0f; }

  int nkv = Ss / 64;
  for (int it = 0; it < nkv; ++it) {
    int cur = it & 1;
    int kv0 = it * 64;
    __syncthreads();  // sVt/sP readers and sK[cur^1] readers (iter-1) done
    if (it + 1 < nkv) kload(kv0 + 64, cur ^ 1);   // prefetch K(i+1)
    {   // V tile, transposed into LDS through VGPRs
      int kv = tid >> 1, d0 = (tid & 1) * 32;
      union { uint4 u[4]; bf16 e[32]; } t;
      const uint4* gv = (const uint4*)(Vg + (size_t)(kv0 + kv) * Dd + d0);
      t.u[0] = gv[0]; t.u[1] = gv[1]; t.u[2] = gv[2]; t.u[3] = gv[3];
#pragma unroll
      for (int d = 0; d < 32; ++d) sVt[(d0 + d) * 64 + kv] = t.e[d];
    }
    if (it + 1 < nkv) { ASYNC_WAIT(4); } else { ASYNC_WAIT(0); }
    __syncthreads();  // K(i) + sVt complete for all waves

    // S = (Q K^T) * scale + maskbias
    v8f sacc[4];
#pragma unroll
    for (int f = 0; f < 4; ++f) {
      v8f z;
#pragma unroll
      for (int r = 0; r < 8; ++r) z[r] = 0.0f;
      v16bf kb0 = load_b_frag(sK[cur] + (f * 16) * 64, 64, 0, lane);
      v16bf kb1 = load_b_frag(sK[cur] + (f * 16) * 64, 64, 32, lane);
      z = wmma_bf16(qa0, kb0, z);
      z = wmma_bf16(qa1, kb1, z);
      float mb = sMask[kv0 + f * 16 + ln];
#pragma unroll
      for (int r = 0; r < 8; ++r) z[r] = z[r] * 0.125f + mb;
      sacc[f] = z;
    }

    // online softmax
    float alpha[8];
#pragma unroll
    for (int r = 0; r < 8; ++r) {
      float v = fmaxf(fmaxf(sacc[0][r], sacc[1][r]), fmaxf(sacc[2][r], sacc[3][r]));
#pragma unroll
      for (int off = 8; off >= 1; off >>= 1) v = fmaxf(v, __shfl_xor(v, off, 16));
      float mn = fmaxf(mrow[r], v);
      alpha[r] = __expf(mrow[r] - mn);
      mrow[r] = mn;
    }
#pragma unroll
    for (int f = 0; f < 4; ++f)
#pragma unroll
      for (int r = 0; r < 8; ++r) sacc[f][r] = __expf(sacc[f][r] - mrow[r]);
#pragma unroll
    for (int r = 0; r < 8; ++r) {
      float v = sacc[0][r] + sacc[1][r] + sacc[2][r] + sacc[3][r];
#pragma unroll
      for (int off = 8; off >= 1; off >>= 1) v += __shfl_xor(v, off, 16);
      lrow[r] = lrow[r] * alpha[r] + v;
    }
#pragma unroll
    for (int j = 0; j < 4; ++j)
#pragma unroll
      for (int r = 0; r < 8; ++r) o[j][r] *= alpha[r];

    // P (C-layout) -> per-wave LDS strip -> A-layout fragments
    // (same-wave DS ops are in-order; no block barrier needed)
    bf16* pw = sP[wave];
#pragma unroll
    for (int f = 0; f < 4; ++f)
#pragma unroll
      for (int r = 0; r < 8; ++r)
        pw[(lh * 8 + r) * 64 + f * 16 + ln] = (bf16)sacc[f][r];

    v16bf pa0 = load_a_frag(pw, 64, lane);        // kv 0..31
    v16bf pa1 = load_a_frag(pw + 32, 64, lane);   // kv 32..63
#pragma unroll
    for (int j = 0; j < 4; ++j) {
      v16bf vb0 = load_b_frag(sVt + (j * 16) * 64, 64, 0, lane);
      v16bf vb1 = load_b_frag(sVt + (j * 16) * 64, 64, 32, lane);
      o[j] = wmma_bf16(pa0, vb0, o[j]);
      o[j] = wmma_bf16(pa1, vb1, o[j]);
    }
  }

  bf16* cg = ctx + (rowbase + (size_t)qt * 64 + wave * 16) * Dd + h * DHh;
#pragma unroll
  for (int j = 0; j < 4; ++j)
#pragma unroll
    for (int r = 0; r < 8; ++r) {
      int m = lh * 8 + r;
      int n = j * 16 + ln;
      cg[(size_t)m * Dd + n] = (bf16)(o[j][r] / lrow[r]);
    }
}

// ---------------------------------------------------------------------------
extern "C" void kernel_launch(void* const* d_in, const int* in_sizes, int n_in,
                              void* d_out, int out_size, void* d_ws, size_t ws_size,
                              hipStream_t stream) {
  const float* q    = (const float*)d_in[0];
  const float* k    = (const float*)d_in[1];
  const float* v    = (const float*)d_in[2];
  const float* mask = (const float*)d_in[3];
  const float* Wq = (const float*)d_in[4];  const float* bq = (const float*)d_in[5];
  const float* Wk = (const float*)d_in[6];  const float* bk = (const float*)d_in[7];
  const float* Wv = (const float*)d_in[8];  const float* bv = (const float*)d_in[9];
  const float* Wo = (const float*)d_in[10]; const float* bo = (const float*)d_in[11];

  char* ws = (char*)d_ws;
  size_t off = 0;
  auto alloc = [&](size_t bytes) {
    void* p = ws + off;
    off += (bytes + 255) & ~(size_t)255;
    return p;
  };
  const size_t actB = (size_t)MROWS * Dd * sizeof(bf16);   // 16 MB
  const size_t wB   = (size_t)Dd * Dd * sizeof(bf16);      // 2 MB

  bf16* qb  = (bf16*)alloc(actB);
  bf16* kb  = (bf16*)alloc(actB);
  bf16* vb  = (bf16*)alloc(actB);
  bf16* Wqt = (bf16*)alloc(wB);
  bf16* Wkt = (bf16*)alloc(wB);
  bf16* Wvt = (bf16*)alloc(wB);
  bf16* Wot = (bf16*)alloc(wB);
  bf16* Qp  = (bf16*)alloc(actB);
  bf16* Kp  = (bf16*)alloc(actB);
  bf16* Vp  = (bf16*)alloc(actB);
  bf16* ctx = (bf16*)alloc(actB);

  long nact = (long)MROWS * Dd;
  dim3 cvtGrid((unsigned)((nact / 4 + 255) / 256));
  cvt_bf16_kernel<<<cvtGrid, 256, 0, stream>>>(q, qb, nact);
  cvt_bf16_kernel<<<cvtGrid, 256, 0, stream>>>(k, kb, nact);
  cvt_bf16_kernel<<<cvtGrid, 256, 0, stream>>>(v, vb, nact);

  dim3 trGrid(Dd / 64, Dd / 64), trBlk(64, 4);
  transpose_bf16_kernel<<<trGrid, trBlk, 0, stream>>>(Wq, Wqt, Dd, Dd);
  transpose_bf16_kernel<<<trGrid, trBlk, 0, stream>>>(Wk, Wkt, Dd, Dd);
  transpose_bf16_kernel<<<trGrid, trBlk, 0, stream>>>(Wv, Wvt, Dd, Dd);
  transpose_bf16_kernel<<<trGrid, trBlk, 0, stream>>>(Wo, Wot, Dd, Dd);

  dim3 gemmGrid(Dd / 128, MROWS / 128);  // (8, 64)
  gemm_bf16_kernel<<<gemmGrid, 128, 0, stream>>>(qb, Wqt, bq, Qp, nullptr, Dd, Dd);
  gemm_bf16_kernel<<<gemmGrid, 128, 0, stream>>>(kb, Wkt, bk, Kp, nullptr, Dd, Dd);
  gemm_bf16_kernel<<<gemmGrid, 128, 0, stream>>>(vb, Wvt, bv, Vp, nullptr, Dd, Dd);

  dim3 faGrid(Ss / 64, Hh, Bb);  // (32, 16, 4)
  flash_attn_kernel<<<faGrid, 128, 0, stream>>>(Qp, Kp, Vp, mask, ctx);

  gemm_bf16_kernel<<<gemmGrid, 128, 0, stream>>>(ctx, Wot, bo, nullptr, (float*)d_out,
                                                 Dd, Dd);
}
